// Block_33285996544097
// MI455X (gfx1250) — compile-verified, matrix operand order checked by reading, not run on hardware
//
#include <hip/hip_runtime.h>
#include <hip/hip_bf16.h>

typedef unsigned short u16;
typedef unsigned int u32;
typedef __attribute__((ext_vector_type(16))) __bf16 v16bf;
typedef __attribute__((ext_vector_type(8)))  float  v8f;
typedef int vs4i __attribute__((vector_size(16)));   // matches builtin param type

union FragBF {
  v16bf v;
  uint4 q[2];
  u16   h[16];
};

__device__ __forceinline__ u16 f32_to_bf16(float f) {
  union { float f; u32 u; } c; c.f = f;
  u32 u = c.u;
  u += 0x7fffu + ((u >> 16) & 1u);   // round-to-nearest-even
  return (u16)(u >> 16);
}

// ---------------------------------------------------------------------------
// Async global->LDS copy (CDNA5 GLOBAL_LOAD_ASYNC_TO_LDS_B128, ASYNCcnt),
// guarded so a toolchain without the builtins still compiles the sync path.
// ---------------------------------------------------------------------------
#if __has_builtin(__builtin_amdgcn_global_load_async_to_lds_b128) && \
    __has_builtin(__builtin_amdgcn_s_wait_asynccnt)
#define USE_ASYNC_LDS 1
#else
#define USE_ASYNC_LDS 0
#endif

#if USE_ASYNC_LDS
__device__ __forceinline__ void cp_async_b128(const u16* g, u16* l) {
  // generic LDS pointer low 32 bits == LDS offset (aperture scheme)
  __builtin_amdgcn_global_load_async_to_lds_b128(
      (__attribute__((address_space(1))) vs4i*)(unsigned long long)(const void*)g,
      (__attribute__((address_space(3))) vs4i*)(unsigned int)(unsigned long long)(void*)l,
      0, 0);
}
__device__ __forceinline__ void wait_async_lds() {
  __builtin_amdgcn_s_wait_asynccnt(0);
}
#endif

// ---------------------------------------------------------------------------
// fp32 -> bf16 cast (weights)
// ---------------------------------------------------------------------------
__global__ __launch_bounds__(256) void cast_bf16_k(const float* __restrict__ in,
                                                   u16* __restrict__ out, int n) {
  int i = blockIdx.x * 256 + threadIdx.x;
  if (i < n) out[i] = f32_to_bf16(in[i]);
}

// ---------------------------------------------------------------------------
// LayerNorm over C=1024, one row per block, bf16 output
// ---------------------------------------------------------------------------
__global__ __launch_bounds__(256) void layernorm_bf16_k(const float* __restrict__ x,
                                                        const float* __restrict__ g,
                                                        const float* __restrict__ b,
                                                        u16* __restrict__ out) {
  constexpr int C = 1024;
  __shared__ float red1[256];
  __shared__ float red2[256];
  const int row = blockIdx.x;
  const int tid = threadIdx.x;
  const float4 v = *(const float4*)&x[(size_t)row * C + tid * 4];
  float s  = v.x + v.y + v.z + v.w;
  float ss = v.x * v.x + v.y * v.y + v.z * v.z + v.w * v.w;
  red1[tid] = s; red2[tid] = ss;
  __syncthreads();
  for (int o = 128; o > 0; o >>= 1) {
    if (tid < o) { red1[tid] += red1[tid + o]; red2[tid] += red2[tid + o]; }
    __syncthreads();
  }
  const float mu  = red1[0] * (1.0f / 1024.0f);
  const float var = red2[0] * (1.0f / 1024.0f) - mu * mu;
  const float rs  = rsqrtf(var + 1e-5f);
#pragma unroll
  for (int j = 0; j < 4; ++j) {
    int c = tid * 4 + j;
    float xv = (&v.x)[j];
    out[(size_t)row * C + c] = f32_to_bf16((xv - mu) * rs * g[c] + b[c]);
  }
}

// ---------------------------------------------------------------------------
// Tiled bf16 GEMM:  Y[M,N] = A[M,K] @ W[N,K]^T  (both bf16, fp32 accum)
// Block tile 128(M) x 128(N), 8 waves (4x2), each wave 32x64 (2x4 subtiles),
// K-step 32, double-buffered LDS, async global->LDS fill when available.
// Fragment layout per CDNA5 ISA 7.12.2: lane l -> row = l%16,
// elements 0..7 -> k = kb..kb+7, elements 8..15 -> k = kb+16..kb+23,
// kb = (l>=16)?8:0.  Row-major A and row-major W both load identically.
// EPI: 0 = bf16 out with q-scale for first 1024 cols (qkv)
//      1 = fp32 out = acc + bias + residual
//      2 = bf16 out = gelu(acc + bias)  (exact erf)
// ---------------------------------------------------------------------------
template <int EPI>
__global__ __launch_bounds__(256) void gemm_bf16_k(
    const u16* __restrict__ A, const u16* __restrict__ W,
    const float* __restrict__ bias, const float* __restrict__ res,
    float* __restrict__ outF, u16* __restrict__ outB,
    int M, int N, int K) {
  __shared__ __align__(16) u16 As[2][128][40];
  __shared__ __align__(16) u16 Ws[2][128][40];

  const int tid  = threadIdx.x;
  const int lane = tid & 31;
  const int wave = tid >> 5;
  const int r    = lane & 15;
  const int hi   = lane >> 4;

  const int m0 = blockIdx.y * 128;
  const int n0 = blockIdx.x * 128;
  const int wm = (wave >> 1) * 32;   // 4 waves along M
  const int wn = (wave & 1) * 64;    // 2 waves along N

  // cooperative copy: each 128x32 tile = 512 16B chunks; thread t owns
  // chunks t and t+256 (rows row, row+64; same col)
  const int row = tid >> 2;
  const int col = (tid & 3) << 3;

  const u16* gA0 = &A[(size_t)(m0 + row) * K + col];
  const u16* gA1 = &A[(size_t)(m0 + row + 64) * K + col];
  const u16* gW0 = &W[(size_t)(n0 + row) * K + col];
  const u16* gW1 = &W[(size_t)(n0 + row + 64) * K + col];

  v8f acc[2][4];
  const v8f zero8 = {0.f, 0.f, 0.f, 0.f, 0.f, 0.f, 0.f, 0.f};
#pragma unroll
  for (int i = 0; i < 2; ++i)
#pragma unroll
    for (int j = 0; j < 4; ++j) acc[i][j] = zero8;

  const int nk = K >> 5;   // K / 32

#if USE_ASYNC_LDS
  // ---- async double-buffered pipeline ----
  cp_async_b128(gA0, &As[0][row][col]);
  cp_async_b128(gA1, &As[0][row + 64][col]);
  cp_async_b128(gW0, &Ws[0][row][col]);
  cp_async_b128(gW1, &Ws[0][row + 64][col]);
  wait_async_lds();
  __syncthreads();
  for (int i = 0; i < nk; ++i) {
    const int cur = i & 1;
    if (i + 1 < nk) {
      const int nxt = cur ^ 1;
      const int kn = (i + 1) << 5;
      cp_async_b128(gA0 + kn, &As[nxt][row][col]);
      cp_async_b128(gA1 + kn, &As[nxt][row + 64][col]);
      cp_async_b128(gW0 + kn, &Ws[nxt][row][col]);
      cp_async_b128(gW1 + kn, &Ws[nxt][row + 64][col]);
    }
    FragBF af[2], bfr[4];
#pragma unroll
    for (int ii = 0; ii < 2; ++ii) {
      int rr = wm + ii * 16 + r;
      af[ii].q[0] = *(const uint4*)&As[cur][rr][hi * 8];
      af[ii].q[1] = *(const uint4*)&As[cur][rr][hi * 8 + 16];
    }
#pragma unroll
    for (int jj = 0; jj < 4; ++jj) {
      int rr = wn + jj * 16 + r;
      bfr[jj].q[0] = *(const uint4*)&Ws[cur][rr][hi * 8];
      bfr[jj].q[1] = *(const uint4*)&Ws[cur][rr][hi * 8 + 16];
    }
#pragma unroll
    for (int ii = 0; ii < 2; ++ii)
#pragma unroll
      for (int jj = 0; jj < 4; ++jj)
        acc[ii][jj] = __builtin_amdgcn_wmma_f32_16x16x32_bf16(
            false, af[ii].v, false, bfr[jj].v, (short)0, acc[ii][jj], false, false);
    wait_async_lds();
    __syncthreads();
  }
#else
  // ---- register-prefetch double-buffered pipeline ----
  uint4 pa0 = *(const uint4*)gA0;
  uint4 pa1 = *(const uint4*)gA1;
  uint4 pw0 = *(const uint4*)gW0;
  uint4 pw1 = *(const uint4*)gW1;
  *(uint4*)&As[0][row][col]      = pa0;
  *(uint4*)&As[0][row + 64][col] = pa1;
  *(uint4*)&Ws[0][row][col]      = pw0;
  *(uint4*)&Ws[0][row + 64][col] = pw1;
  __syncthreads();
  for (int i = 0; i < nk; ++i) {
    const int cur = i & 1;
    const bool more = (i + 1 < nk);
    if (more) {
      const int kn = (i + 1) << 5;
      pa0 = *(const uint4*)(gA0 + kn);
      pa1 = *(const uint4*)(gA1 + kn);
      pw0 = *(const uint4*)(gW0 + kn);
      pw1 = *(const uint4*)(gW1 + kn);
    }
    FragBF af[2], bfr[4];
#pragma unroll
    for (int ii = 0; ii < 2; ++ii) {
      int rr = wm + ii * 16 + r;
      af[ii].q[0] = *(const uint4*)&As[cur][rr][hi * 8];
      af[ii].q[1] = *(const uint4*)&As[cur][rr][hi * 8 + 16];
    }
#pragma unroll
    for (int jj = 0; jj < 4; ++jj) {
      int rr = wn + jj * 16 + r;
      bfr[jj].q[0] = *(const uint4*)&Ws[cur][rr][hi * 8];
      bfr[jj].q[1] = *(const uint4*)&Ws[cur][rr][hi * 8 + 16];
    }
#pragma unroll
    for (int ii = 0; ii < 2; ++ii)
#pragma unroll
      for (int jj = 0; jj < 4; ++jj)
        acc[ii][jj] = __builtin_amdgcn_wmma_f32_16x16x32_bf16(
            false, af[ii].v, false, bfr[jj].v, (short)0, acc[ii][jj], false, false);
    if (more) {
      const int nxt = cur ^ 1;
      *(uint4*)&As[nxt][row][col]      = pa0;
      *(uint4*)&As[nxt][row + 64][col] = pa1;
      *(uint4*)&Ws[nxt][row][col]      = pw0;
      *(uint4*)&Ws[nxt][row + 64][col] = pw1;
    }
    __syncthreads();
  }
#endif

  // epilogue: D layout lane mapping m = v + 8*hi, n = lane%16
#pragma unroll
  for (int i = 0; i < 2; ++i)
#pragma unroll
    for (int j = 0; j < 4; ++j) {
#pragma unroll
      for (int v = 0; v < 8; ++v) {
        int m = m0 + wm + i * 16 + v + 8 * hi;
        int n = n0 + wn + j * 16 + r;
        float val = acc[i][j][v];
        size_t idx = (size_t)m * N + n;
        if constexpr (EPI == 0) {
          float s = (n < 1024) ? 0.125f : 1.0f;   // SCALE = 64^-0.5 on Q
          outB[idx] = f32_to_bf16(val * s);
        } else if constexpr (EPI == 1) {
          outF[idx] = val + bias[n] + res[idx];
        } else {
          float t = val + bias[n];
          float gl = 0.5f * t * (1.0f + erff(t * 0.70710678118654752f));
          outB[idx] = f32_to_bf16(gl);
        }
      }
    }
}

// ---------------------------------------------------------------------------
// V transpose: vT[b,h,d,n] = qkv[b*N+n, 2C + h*64 + d]   (bf16)
// ---------------------------------------------------------------------------
__global__ __launch_bounds__(256) void transpose_v_k(const u16* __restrict__ qkv,
                                                     u16* __restrict__ vT) {
  int i = blockIdx.x * 256 + threadIdx.x;     // 0 .. 8M-1
  int n  = i & 2047;
  int d  = (i >> 11) & 63;
  int bh = i >> 17;
  int b  = bh >> 4, h = bh & 15;
  vT[i] = qkv[((size_t)(b * 2048 + n)) * 3072 + 2048 + h * 64 + d];
}

// ---------------------------------------------------------------------------
// Flash attention: one (b,h) x 128-query block per workgroup, 16 q-rows/wave.
// S = Q@K^T via 2 WMMAs per 16-key tile; online softmax with shfl_xor row
// reductions (wave32, masks 1/2/4/8 stay inside each 16-lane half = one row);
// P goes through LDS to convert D-layout -> A-layout; O += P@V via 4 WMMAs.
// ---------------------------------------------------------------------------
__global__ __launch_bounds__(256) void flash_attn_k(const u16* __restrict__ qkv,
                                                    const u16* __restrict__ vT,
                                                    u16* __restrict__ o) {
  constexpr int Nctx = 2048, C = 1024, C3 = 3072, Dh = 64;
  __shared__ __align__(16) u16 Ps[8][16][32];

  const int tid  = threadIdx.x;
  const int lane = tid & 31;
  const int wave = tid >> 5;
  const int r    = lane & 15;
  const int hi   = lane >> 4;

  const int bh = blockIdx.y;   // b*16 + h
  const int b  = bh >> 4;
  const int h  = bh & 15;
  const int q0 = blockIdx.x * 128 + wave * 16;

  const v8f zero8 = {0.f, 0.f, 0.f, 0.f, 0.f, 0.f, 0.f, 0.f};

  // Q fragments (A-layout), Q already scaled by 0.125 in qkv epilogue
  FragBF qa[2];
  {
    const u16* qrow = qkv + (size_t)(b * Nctx + q0 + r) * C3 + h * Dh;
#pragma unroll
    for (int f = 0; f < 2; ++f) {
      int d = f * 32 + hi * 8;
      qa[f].q[0] = *(const uint4*)&qrow[d];
      qa[f].q[1] = *(const uint4*)&qrow[d + 16];
    }
  }

  float mrow[8], lrow[8];
  v8f oacc[4];
#pragma unroll
  for (int v = 0; v < 8; ++v) { mrow[v] = -1e30f; lrow[v] = 0.f; }
#pragma unroll
  for (int d = 0; d < 4; ++d) oacc[d] = zero8;

  for (int kb = 0; kb < Nctx; kb += 32) {
    v8f s[2];
#pragma unroll
    for (int t = 0; t < 2; ++t) {
      const u16* krow = qkv + (size_t)(b * Nctx + kb + t * 16 + r) * C3 + C + h * Dh;
      v8f st = zero8;
#pragma unroll
      for (int f = 0; f < 2; ++f) {
        FragBF kf;
        int d = f * 32 + hi * 8;
        kf.q[0] = *(const uint4*)&krow[d];
        kf.q[1] = *(const uint4*)&krow[d + 16];
        st = __builtin_amdgcn_wmma_f32_16x16x32_bf16(
            false, qa[f].v, false, kf.v, (short)0, st, false, false);
      }
      s[t] = st;
    }

    // online softmax: per VGPR v, row = v + 8*hi
#pragma unroll
    for (int v = 0; v < 8; ++v) {
      float rm = fmaxf(s[0][v], s[1][v]);
#pragma unroll
      for (int off = 1; off < 16; off <<= 1)
        rm = fmaxf(rm, __shfl_xor(rm, off, 32));
      float mnew = fmaxf(mrow[v], rm);
      float sc   = __expf(mrow[v] - mnew);
      mrow[v] = mnew;
      lrow[v] *= sc;
#pragma unroll
      for (int d = 0; d < 4; ++d) oacc[d][v] *= sc;
      float p0 = __expf(s[0][v] - mnew);
      float p1 = __expf(s[1][v] - mnew);
      float ps = p0 + p1;
#pragma unroll
      for (int off = 1; off < 16; off <<= 1)
        ps += __shfl_xor(ps, off, 32);
      lrow[v] += ps;
      s[0][v] = p0;
      s[1][v] = p1;
    }

    __syncthreads();   // previous iteration's P reads complete
#pragma unroll
    for (int v = 0; v < 8; ++v) {
      Ps[wave][v + 8 * hi][r]      = f32_to_bf16(s[0][v]);
      Ps[wave][v + 8 * hi][16 + r] = f32_to_bf16(s[1][v]);
    }
    __syncthreads();   // P visible for A-layout reads

    FragBF pa;
    pa.q[0] = *(const uint4*)&Ps[wave][r][hi * 8];
    pa.q[1] = *(const uint4*)&Ps[wave][r][hi * 8 + 16];
#pragma unroll
    for (int ds = 0; ds < 4; ++ds) {
      const u16* vrow = vT + ((size_t)bh * Dh + ds * 16 + r) * Nctx + kb;
      FragBF vf;
      vf.q[0] = *(const uint4*)&vrow[hi * 8];
      vf.q[1] = *(const uint4*)&vrow[hi * 8 + 16];
      oacc[ds] = __builtin_amdgcn_wmma_f32_16x16x32_bf16(
          false, pa.v, false, vf.v, (short)0, oacc[ds], false, false);
    }
  }

  // normalize and write O as [M, C] bf16 (col = h*64 + d)
#pragma unroll
  for (int v = 0; v < 8; ++v) {
    float inv = 1.0f / lrow[v];
    int m = b * Nctx + q0 + v + 8 * hi;
#pragma unroll
    for (int ds = 0; ds < 4; ++ds)
      o[(size_t)m * C + h * Dh + ds * 16 + r] = f32_to_bf16(oacc[ds][v] * inv);
  }
}

// ---------------------------------------------------------------------------
// host-side orchestration
// ---------------------------------------------------------------------------
extern "C" void kernel_launch(void* const* d_in, const int* in_sizes, int n_in,
                              void* d_out, int out_size, void* d_ws, size_t ws_size,
                              hipStream_t stream) {
  const float* x      = (const float*)d_in[0];
  const float* ln1_g  = (const float*)d_in[1];
  const float* ln1_b  = (const float*)d_in[2];
  const float* qkv_w  = (const float*)d_in[3];
  const float* proj_w = (const float*)d_in[4];
  const float* proj_b = (const float*)d_in[5];
  const float* ln2_g  = (const float*)d_in[6];
  const float* ln2_b  = (const float*)d_in[7];
  const float* fc1_w  = (const float*)d_in[8];
  const float* fc1_b  = (const float*)d_in[9];
  const float* fc2_w  = (const float*)d_in[10];
  const float* fc2_b  = (const float*)d_in[11];
  float* out = (float*)d_out;

  const int M = 8192, C = 1024, C3 = 3072, Hd = 4096;

  char* ws = (char*)d_ws;
  u16*   h_bf   = (u16*)(ws);                             // 16 MB  (LN out, reused)
  u16*   qkv_bf = (u16*)(ws + (size_t)16  * 1048576);     // 48 MB
  u16*   vTb    = (u16*)(ws + (size_t)64  * 1048576);     // 16 MB
  u16*   o_bf   = (u16*)(ws + (size_t)80  * 1048576);     // 16 MB
  float* x1     = (float*)(ws + (size_t)96 * 1048576);    // 32 MB
  u16*   h2_bf  = (u16*)(ws + (size_t)128 * 1048576);     // 64 MB
  u16*   wqkv   = (u16*)(ws + (size_t)192 * 1048576);     // 24 MB total weights
  u16*   wproj  = wqkv  + (size_t)C3 * C;
  u16*   wfc1   = wproj + (size_t)C  * C;
  u16*   wfc2   = wfc1  + (size_t)Hd * C;

  // weights -> bf16
  cast_bf16_k<<<(C3 * C) / 256, 256, 0, stream>>>(qkv_w,  wqkv,  C3 * C);
  cast_bf16_k<<<(C  * C) / 256, 256, 0, stream>>>(proj_w, wproj, C * C);
  cast_bf16_k<<<(Hd * C) / 256, 256, 0, stream>>>(fc1_w,  wfc1,  Hd * C);
  cast_bf16_k<<<(C * Hd) / 256, 256, 0, stream>>>(fc2_w,  wfc2,  C * Hd);

  // attention branch
  layernorm_bf16_k<<<M, 256, 0, stream>>>(x, ln1_g, ln1_b, h_bf);
  gemm_bf16_k<0><<<dim3(C3 / 128, M / 128), 256, 0, stream>>>(
      h_bf, wqkv, nullptr, nullptr, nullptr, qkv_bf, M, C3, C);
  transpose_v_k<<<(4 * 16 * 64 * 2048) / 256, 256, 0, stream>>>(qkv_bf, vTb);
  flash_attn_k<<<dim3(2048 / 128, 4 * 16), 256, 0, stream>>>(qkv_bf, vTb, o_bf);
  gemm_bf16_k<1><<<dim3(C / 128, M / 128), 256, 0, stream>>>(
      o_bf, wproj, proj_b, x, x1, nullptr, M, C, C);

  // MLP branch
  layernorm_bf16_k<<<M, 256, 0, stream>>>(x1, ln2_g, ln2_b, h_bf);
  gemm_bf16_k<2><<<dim3(Hd / 128, M / 128), 256, 0, stream>>>(
      h_bf, wfc1, fc1_b, nullptr, nullptr, h2_bf, M, Hd, C);
  gemm_bf16_k<1><<<dim3(C / 128, M / 128), 256, 0, stream>>>(
      h2_bf, wfc2, fc2_b, x1, out, nullptr, M, C, Hd);
}
